// SupervisedCL_43602507989898
// MI455X (gfx1250) — matrix operand order repaired and checked by reading, not compiled, not run on hardware
//
#include <hip/hip_runtime.h>
#include <hip/hip_bf16.h>
#include <math.h>

// ---------------- problem constants (from reference) ----------------
#define BATCH    256
#define NCLS     20
#define KCLS     10
#define NK       200        // NCLS * KCLS anchors per batch element
#define DIM      512
#define TINV     10.0f      // 1 / T
#define ALPHA    0.1f
#define BNORM    1.0f
#define ROWTILES 13         // ceil(200 / 16)
#define PADROWS  208        // ROWTILES * 16 (zero-padded rows 200..207)

typedef __attribute__((ext_vector_type(16))) _Float16 v16h;
typedef __attribute__((ext_vector_type(8)))  float    v8f;

union F16Frag { v16h v; unsigned int u[8]; };

// ---------------------------------------------------------------
// Kernel 1: L2-normalize each 512-vector, store as f16 into a
// row-padded [BATCH][208][512] layout (rows 200..207 zeroed) so every
// 16-row tile is one contiguous 16 KB block.
// ---------------------------------------------------------------
__global__ __launch_bounds__(128)
void SCL_norm_kernel(const float* __restrict__ op, _Float16* __restrict__ pn) {
    const int vec = blockIdx.x;                      // 0 .. BATCH*PADROWS-1
    const int b   = vec / PADROWS;
    const int r   = vec % PADROWS;
    const int t   = threadIdx.x;                     // 0 .. 127

    _Float16* dst = pn + (size_t)vec * DIM + t * 4;
    if (r >= NK) {                                   // zero padding rows
        *(unsigned long long*)dst = 0ull;
        return;
    }

    const float4* src = (const float4*)(op + ((size_t)b * NK + r) * DIM);
    float4 f = src[t];

    float s = f.x * f.x + f.y * f.y + f.z * f.z + f.w * f.w;
    #pragma unroll
    for (int off = 16; off >= 1; off >>= 1)
        s += __shfl_xor(s, off, 32);

    __shared__ float sdata[4];
    const int wave = t >> 5;
    if ((t & 31) == 0) sdata[wave] = s;
    __syncthreads();
    const float total = sdata[0] + sdata[1] + sdata[2] + sdata[3];
    const float rinv  = rsqrtf(total);

    _Float16 h0 = (_Float16)(f.x * rinv);
    _Float16 h1 = (_Float16)(f.y * rinv);
    _Float16 h2 = (_Float16)(f.z * rinv);
    _Float16 h3 = (_Float16)(f.w * rinv);
    unsigned long long pack =
        ((unsigned long long)__builtin_bit_cast(unsigned short, h0)) |
        ((unsigned long long)__builtin_bit_cast(unsigned short, h1) << 16) |
        ((unsigned long long)__builtin_bit_cast(unsigned short, h2) << 32) |
        ((unsigned long long)__builtin_bit_cast(unsigned short, h3) << 48);
    *(unsigned long long*)dst = pack;
}

// ---------------------------------------------------------------
// Async copy of one contiguous 16 KB tile (16 rows x 512 f16) from the
// padded P matrix into LDS, bypassing VGPRs: 32 issues of
// global_load_async_to_lds_b128 (512 B per issue across the wave32).
// The LDS address is derived from the real shared-memory pointer
// (ptrtoint => pointer is captured, so the "memory" clobbers make the
// compiler honor the asm's LDS writes instead of hoisting LDS reads).
// ---------------------------------------------------------------
#define TILE_BYTES (16u * DIM * 2u)   // 16384

__device__ __forceinline__
void async_tile_load(unsigned long long gbase, int tileRowBase,
                     _Float16* ldsDst, int lane) {
    const unsigned gtile   = (unsigned)tileRowBase * (DIM * 2u); // byte off in batch
    const unsigned ldsBase = (unsigned)(uintptr_t)ldsDst;        // low 32 = LDS offset
    #pragma unroll
    for (int k = 0; k < 32; ++k) {
        const unsigned chunk = (unsigned)(k * 32 + lane) * 16u;
        const unsigned ldo   = ldsBase + chunk;   // per-lane LDS byte address
        const unsigned vgo   = gtile + chunk;     // per-lane global byte offset
        asm volatile("global_load_async_to_lds_b128 %0, %1, %2"
                     :: "v"(ldo), "v"(vgo), "s"(gbase)
                     : "memory");
    }
}

// ---------------------------------------------------------------
// Kernel 2: per (batch, row-tile-of-16) compute a 16x208 slice of the
// Gram matrix with v_wmma_f32_16x16x32_f16, double-buffering B tiles
// through the CDNA5 async-to-LDS path, folding each 16x16 tile straight
// into exp-sum / pos-sum / self-sim row accumulators. One wave32/block.
// ---------------------------------------------------------------
__global__ __launch_bounds__(32)
void SCL_gram_loss_kernel(const _Float16* __restrict__ P, float* __restrict__ partial) {
    // [0,16K) = A tile, [16K,32K) = B buf0, [32K,48K) = B buf1.
    __shared__ _Float16 smem[3 * 16 * DIM];

    const int bid     = blockIdx.x;
    const int b       = bid / ROWTILES;
    const int rt      = bid % ROWTILES;
    const int rowBase = rt * 16;
    const int lane    = threadIdx.x;          // 0..31 (wave32)
    const int m       = lane & 15;
    const int half    = lane >> 4;

    const unsigned long long Pb =
        (unsigned long long)(uintptr_t)(P + (size_t)b * PADROWS * DIM);

    // Prologue: async-load A row-tile and B col-tile 0.
    async_tile_load(Pb, rowBase, smem,            lane);   // A
    async_tile_load(Pb, 0,       smem + 16 * DIM, lane);   // B tile 0 -> buf0

    float expAcc[8], posAcc[8], selfV[8];
    #pragma unroll
    for (int r = 0; r < 8; ++r) { expAcc[r] = 0.0f; posAcc[r] = 0.0f; selfV[r] = 0.0f; }

    for (int ct = 0; ct < ROWTILES; ++ct) {
        const int buf = ct & 1;

        // Issue next B tile into the other buffer, then wait until all
        // previously issued tiles have landed (async loads complete in
        // order, so <=32 outstanding == everything before the last 32 done).
        if (ct + 1 < ROWTILES) {
            async_tile_load(Pb, (ct + 1) * 16,
                            smem + (1 + ((ct + 1) & 1)) * 16 * DIM, lane);
            asm volatile("s_wait_asynccnt 0x20" ::: "memory");
        } else {
            asm volatile("s_wait_asynccnt 0x0" ::: "memory");
        }

        const _Float16* At = smem;                        // A tile
        const _Float16* Bt = smem + (1 + buf) * 16 * DIM; // current B tile

        // ---- 16x16 tile of sims via WMMA over K = 512 (16 chunks of 32) ----
        v8f acc = {};
        #pragma unroll
        for (int kc = 0; kc < DIM / 32; ++kc) {
            const int K0 = kc * 32;
            F16Frag af, bf;
            #pragma unroll
            for (int vi = 0; vi < 8; ++vi) {
                // A 16x32 f16 layout: M = lane%16; dwords 0-3 hold K pairs
                // {2v,2v+1}+8*half, dwords 4-7 hold {16+2(v-4),...}+8*half.
                const int ka = K0 + ((vi < 4) ? (2 * vi + 8 * half)
                                              : (16 + 2 * (vi - 4) + 8 * half));
                af.u[vi] = *(const unsigned int*)&At[m * DIM + ka];
                // B 32x16 f16 layout: N = lane%16; same K order as A
                // (dwords 0-3 K {2v,2v+1}+8*half, dwords 4-7 {16+2(v-4),...}+8*half).
                const int kb = ka;
                bf.u[vi] = *(const unsigned int*)&Bt[m * DIM + kb];
            }
            acc = __builtin_amdgcn_wmma_f32_16x16x32_f16(
                false, af.v, false, bf.v, (short)0, acc, false, false);
        }

        // ---- fold tile into per-row accumulators ----
        const int  gcol     = ct * 16 + m;          // this lane's column
        const bool colValid = (gcol < NK);
        const int  colCls   = gcol / KCLS;
        #pragma unroll
        for (int r = 0; r < 8; ++r) {
            const int   grow = rowBase + r + 8 * half;   // row for acc[r]
            const float s    = acc[r] * TINV;            // sim / T
            if (colValid) {
                expAcc[r] += __expf(s);
                if (colCls == grow / KCLS) posAcc[r] += s;
                if (gcol == grow)          selfV[r]  = s;
            }
        }
    }

    // ---- reduce across the 16 lanes sharing each row set ----
    #pragma unroll
    for (int r = 0; r < 8; ++r) {
        #pragma unroll
        for (int off = 1; off < 16; off <<= 1) {
            expAcc[r] += __shfl_xor(expAcc[r], off, 16);
            posAcc[r] += __shfl_xor(posAcc[r], off, 16);
            selfV[r]  += __shfl_xor(selfV[r],  off, 16);
        }
    }

    float local = 0.0f;
    if (m == 0) {   // lanes 0 (rows rowBase..+7) and 16 (rows rowBase+8..+15)
        #pragma unroll
        for (int r = 0; r < 8; ++r) {
            const int grow = rowBase + r + 8 * half;
            if (grow < NK) {
                const float exp_term = expAcc[r] - __expf(selfV[r]);
                const float log_term = __logf(exp_term);
                const float sum_term = (posAcc[r] - selfV[r]) / (float)(KCLS - 1);
                local += log_term - sum_term;
            }
        }
    }
    local += __shfl_xor(local, 16, 32);   // combine lane 0 + lane 16
    if (lane == 0) partial[bid] = local;
}

// ---------------------------------------------------------------
// Kernel 3: deterministic single-block reduction of partials -> scalar.
// ---------------------------------------------------------------
__global__ __launch_bounds__(256)
void SCL_reduce_kernel(const float* __restrict__ partial, int n, float* __restrict__ out) {
    const int t = threadIdx.x;
    float s = 0.0f;
    for (int i = t; i < n; i += 256) s += partial[i];
    #pragma unroll
    for (int off = 16; off >= 1; off >>= 1)
        s += __shfl_xor(s, off, 32);
    __shared__ float sdata[8];
    if ((t & 31) == 0) sdata[t >> 5] = s;
    __syncthreads();
    if (t == 0) {
        float tot = 0.0f;
        #pragma unroll
        for (int w = 0; w < 8; ++w) tot += sdata[w];
        out[0] = ALPHA * tot / (BNORM * (float)NCLS * (float)KCLS * (float)DIM);
    }
}

// ---------------------------------------------------------------
extern "C" void kernel_launch(void* const* d_in, const int* in_sizes, int n_in,
                              void* d_out, int out_size, void* d_ws, size_t ws_size,
                              hipStream_t stream) {
    const float* op = (const float*)d_in[0];

    _Float16* Pn      = (_Float16*)d_ws;   // BATCH*PADROWS*DIM f16 (~54.5 MB)
    float*    partial = (float*)((char*)d_ws +
                                 (size_t)BATCH * PADROWS * DIM * sizeof(_Float16));

    SCL_norm_kernel<<<BATCH * PADROWS, 128, 0, stream>>>(op, Pn);
    SCL_gram_loss_kernel<<<BATCH * ROWTILES, 32, 0, stream>>>(Pn, partial);
    SCL_reduce_kernel<<<1, 256, 0, stream>>>(partial, BATCH * ROWTILES, (float*)d_out);
}
